// SegLSTMWithZoneout_83648783057089
// MI455X (gfx1250) — compile-verified, matrix operand order checked by reading, not run on hardware
//
#include <hip/hip_runtime.h>
#include <hip/hip_bf16.h>

// ---------------------------------------------------------------------------
// SegLSTMWithZoneout for MI455X (gfx1250), wave32 + bf16 WMMA.
//
// Phase 1: xg[t,b,g] = x[t,b,:] @ W_ih[g,:]^T + (b_ih+b_hh)[g]  (parallel GEMM),
//          software-pipelined A/B fragments, output in raw WMMA C-fragment
//          layout [t][ntile][mtile][lane][8].
// Phase 2: persistent 32-block scan; W_hh fragments live in VGPRs across all T
//          steps; A (h) fragments double-buffered; grid barrier per step.
// Phase 3: segout gather via reset_idx.
// ---------------------------------------------------------------------------

typedef __attribute__((ext_vector_type(16))) __bf16 v16bf;
typedef __attribute__((ext_vector_type(8)))  float  v8f;
typedef __attribute__((ext_vector_type(4)))  float  v4f;

union BFVec { v16bf v; uint4 u[2]; };

#define HDIM  512
#define BDIM  64
#define GDIM  2048   // 4*H
#define NSEG  128
#define NBLK  32     // persistent blocks in the scan kernel

// ---------------------------------------------------------------------------
// Precompute reset/valid info from durations. durs laid out (S, B) row-major.
__global__ void prep_k(const int* __restrict__ durs,
                       int* __restrict__ reset_idx,      // (B,S)
                       int* __restrict__ mel_len,        // (B)
                       unsigned char* __restrict__ resetflag) // (T,B), pre-zeroed
{
    int b = threadIdx.x;
    if (b >= BDIM) return;
    int acc = 0;
    for (int s = 0; s < NSEG; ++s) {
        acc += durs[s * BDIM + b];
        reset_idx[b * NSEG + s] = acc - 1;
        resetflag[(acc - 1) * BDIM + b] = 1;
    }
    mel_len[b] = acc;
}

// ---------------------------------------------------------------------------
// Convert weights fp32 -> bf16; fuse the two bias vectors.
__global__ void cvt_k(const float* __restrict__ Wih, const float* __restrict__ Whh,
                      const float* __restrict__ bih, const float* __restrict__ bhh,
                      unsigned short* __restrict__ WihB_, unsigned short* __restrict__ WhhB_,
                      float* __restrict__ biassum)
{
    __bf16* WihB = (__bf16*)WihB_;
    __bf16* WhhB = (__bf16*)WhhB_;
    int i = blockIdx.x * 256 + threadIdx.x;   // grid covers 2048*512
    WihB[i] = (__bf16)Wih[i];
    WhhB[i] = (__bf16)Whh[i];
    if (i < GDIM) biassum[i] = bih[i] + bhh[i];
}

// ---------------------------------------------------------------------------
// Phase-1 GEMM: xg = x @ Wih^T + biassum.  grid = (T*16), block = 256 (8 waves).
// Block (t, nblk) computes gate cols nblk*128 .. +127 for all 64 batch rows.
// Wave w: mtile = w&3, ntiles (w>>2)*4 .. +3.  A raw floats and B fragments are
// double-buffered one kb ahead so loads overlap the 4 WMMAs of the current kb.
__global__ __launch_bounds__(256) void gemm_xg_k(
    const float* __restrict__ x,
    const unsigned short* __restrict__ WihB_,
    const float* __restrict__ biassum,
    float* __restrict__ xg)
{
    const __bf16* Wih = (const __bf16*)WihB_;
    int bx    = blockIdx.x;
    int t     = bx >> 4;
    int nblk  = bx & 15;
    int w     = threadIdx.x >> 5;
    int lane  = threadIdx.x & 31;
    int mtile = w & 3;
    int lrow  = lane & 15;
    int lhalf = lane >> 4;
    int koff  = lhalf * 8;
    int gbase = nblk * 128 + (w >> 2) * 64;   // 4 ntiles of 16 from here

    v8f acc[4];
#pragma unroll
    for (int nt = 0; nt < 4; ++nt) {
        float bs = biassum[gbase + nt * 16 + lrow];
#pragma unroll
        for (int r = 0; r < 8; ++r) acc[nt][r] = bs;
    }

    const float*  xrow = x + (size_t)(t * BDIM + mtile * 16 + lrow) * HDIM + koff;
    const __bf16* wrow = Wih + (size_t)(gbase + lrow) * HDIM + koff;

    float4 fr[2][4];
    BFVec  bvb[2][4];
    // Prologue: load kb=0 operands.
    fr[0][0] = *(const float4*)(xrow);
    fr[0][1] = *(const float4*)(xrow + 4);
    fr[0][2] = *(const float4*)(xrow + 16);
    fr[0][3] = *(const float4*)(xrow + 20);
#pragma unroll
    for (int nt = 0; nt < 4; ++nt) {
        const __bf16* wp = wrow + (size_t)nt * 16 * HDIM;
        bvb[0][nt].u[0] = *(const uint4*)(wp);
        bvb[0][nt].u[1] = *(const uint4*)(wp + 16);
    }
#pragma unroll
    for (int kb = 0; kb < 16; ++kb) {
        int cur = kb & 1, nxt = cur ^ 1;
        if (kb < 15) {
            int k1 = (kb + 1) * 32;
            fr[nxt][0] = *(const float4*)(xrow + k1);
            fr[nxt][1] = *(const float4*)(xrow + k1 + 4);
            fr[nxt][2] = *(const float4*)(xrow + k1 + 16);
            fr[nxt][3] = *(const float4*)(xrow + k1 + 20);
#pragma unroll
            for (int nt = 0; nt < 4; ++nt) {
                const __bf16* wp = wrow + (size_t)nt * 16 * HDIM + k1;
                bvb[nxt][nt].u[0] = *(const uint4*)(wp);
                bvb[nxt][nt].u[1] = *(const uint4*)(wp + 16);
            }
        }
        // A fragment: fp32 -> bf16 in-register (ISA 16-bit A layout, 16x32)
        v16bf a;
        a[0]=(__bf16)fr[cur][0].x;  a[1]=(__bf16)fr[cur][0].y;
        a[2]=(__bf16)fr[cur][0].z;  a[3]=(__bf16)fr[cur][0].w;
        a[4]=(__bf16)fr[cur][1].x;  a[5]=(__bf16)fr[cur][1].y;
        a[6]=(__bf16)fr[cur][1].z;  a[7]=(__bf16)fr[cur][1].w;
        a[8]=(__bf16)fr[cur][2].x;  a[9]=(__bf16)fr[cur][2].y;
        a[10]=(__bf16)fr[cur][2].z; a[11]=(__bf16)fr[cur][2].w;
        a[12]=(__bf16)fr[cur][3].x; a[13]=(__bf16)fr[cur][3].y;
        a[14]=(__bf16)fr[cur][3].z; a[15]=(__bf16)fr[cur][3].w;
#pragma unroll
        for (int nt = 0; nt < 4; ++nt) {
            acc[nt] = __builtin_amdgcn_wmma_f32_16x16x32_bf16(
                          false, a, false, bvb[cur][nt].v, (short)0, acc[nt], false, false);
        }
    }
    // Store raw fragments, lane-contiguous (fully coalesced b128 stores).
#pragma unroll
    for (int nt = 0; nt < 4; ++nt) {
        int ntile = nblk * 8 + (w >> 2) * 4 + nt;
        float* tp = xg + (((size_t)t * 128 + ntile) * 4 + mtile) * 256 + lane * 8;
        v4f lo = { acc[nt][0], acc[nt][1], acc[nt][2], acc[nt][3] };
        v4f hi = { acc[nt][4], acc[nt][5], acc[nt][6], acc[nt][7] };
        *(v4f*)(tp)     = lo;
        *(v4f*)(tp + 4) = hi;
    }
}

// ---------------------------------------------------------------------------
// Phase-2 persistent scan. NBLK=32 blocks, 256 threads (8 waves).
// Block nb owns h columns [nb*16, nb*16+16) -> gate cols q*512+nb*16+j, q=0..3.
// Wave w: gate q = w&3, mtiles {2*(w>>2), 2*(w>>2)+1}.
// W_hh fragments for gate q (16 kb-blocks) are preloaded into VGPRs and reused
// for every time step; only h (A) fragments are loaded in-loop, double-buffered.
__global__ __launch_bounds__(256) void seg_lstm_k(
    const float* __restrict__ xg,              // fragment layout
    const unsigned short* __restrict__ WhhB_,
    unsigned short* __restrict__ hbuf_,        // 2 * B*H bf16, buf0 pre-zeroed
    const unsigned char* __restrict__ resetflag,
    const int* __restrict__ mel_len,
    float* __restrict__ outputs,               // (T,B,H)
    float* __restrict__ h_out,
    float* __restrict__ c_out,
    int* __restrict__ counter, int T)
{
    const __bf16* Whh = (const __bf16*)WhhB_;
    __bf16* hbuf = (__bf16*)hbuf_;

    __shared__ float gates[4 * 64 * 16];   // [q][row][col]
    __shared__ float hstate[64 * 16];
    __shared__ float cstate[64 * 16];

    int nb   = blockIdx.x;
    int hc0  = nb * 16;
    int tid  = threadIdx.x;
    int w    = tid >> 5, lane = tid & 31;
    int q    = w & 3;
    int mh   = w >> 2;                 // mtiles 2*mh, 2*mh+1
    int lrow = lane & 15, lhalf = lane >> 4, koff = lhalf * 8;

    // ---- Preload all 16 B fragments for this wave's gate (VGPR-resident) ----
    BFVec bfr[16];
    {
        const __bf16* wrow = Whh + (size_t)(q * HDIM + hc0 + lrow) * HDIM + koff;
#pragma unroll
        for (int kb = 0; kb < 16; ++kb) {
            bfr[kb].u[0] = *(const uint4*)(wrow + kb * 32);
            bfr[kb].u[1] = *(const uint4*)(wrow + kb * 32 + 16);
        }
    }

    for (int i = tid; i < 64 * 16; i += 256) { hstate[i] = 0.f; cstate[i] = 0.f; }

    int eb = tid >> 2;                 // elementwise: batch row
    int j0 = (tid & 3) * 4;            // elementwise: 4 local cols
    int mel = mel_len[eb];
    __syncthreads();

    for (int t = 0; t < T; ++t) {
        const __bf16* hcur = hbuf + (size_t)(t & 1) * (BDIM * HDIM);
        __bf16*       hnxt = hbuf + (size_t)((t + 1) & 1) * (BDIM * HDIM);

        // Accumulator init: coalesced fragment loads of the xg slice.
        v8f acc[2];
#pragma unroll
        for (int mm = 0; mm < 2; ++mm) {
            int m = mh * 2 + mm;
            const float* tp = xg + (((size_t)t * 128 + (q * 32 + nb)) * 4 + m) * 256 + lane * 8;
            v4f lo = *(const v4f*)(tp);
            v4f hi = *(const v4f*)(tp + 4);
            acc[mm][0]=lo.x; acc[mm][1]=lo.y; acc[mm][2]=lo.z; acc[mm][3]=lo.w;
            acc[mm][4]=hi.x; acc[mm][5]=hi.y; acc[mm][6]=hi.z; acc[mm][7]=hi.w;
        }

        // Prefetch next step's xg slice (16KB/block) while this step computes.
        if (t + 1 < T) {
            int pq   = tid >> 6;              // 0..3 : gate region
            int poff = (tid & 63) * 16;       // 64B per thread
            const float* pp = xg + ((size_t)(t + 1) * 128 + (pq * 32 + nb)) * 4 * 256 + poff;
            __builtin_prefetch(pp, 0, 1);
        }

        // ---- K loop: B in registers, A double-buffered ----
        const __bf16* arow0 = hcur + (size_t)((mh * 2 + 0) * 16 + lrow) * HDIM + koff;
        const __bf16* arow1 = hcur + (size_t)((mh * 2 + 1) * 16 + lrow) * HDIM + koff;
        BFVec af[2][2];
        af[0][0].u[0] = *(const uint4*)(arow0);
        af[0][0].u[1] = *(const uint4*)(arow0 + 16);
        af[0][1].u[0] = *(const uint4*)(arow1);
        af[0][1].u[1] = *(const uint4*)(arow1 + 16);
#pragma unroll
        for (int kb = 0; kb < 16; ++kb) {
            int cur = kb & 1, nxt = cur ^ 1;
            if (kb < 15) {
                af[nxt][0].u[0] = *(const uint4*)(arow0 + (kb + 1) * 32);
                af[nxt][0].u[1] = *(const uint4*)(arow0 + (kb + 1) * 32 + 16);
                af[nxt][1].u[0] = *(const uint4*)(arow1 + (kb + 1) * 32);
                af[nxt][1].u[1] = *(const uint4*)(arow1 + (kb + 1) * 32 + 16);
            }
            acc[0] = __builtin_amdgcn_wmma_f32_16x16x32_bf16(
                         false, af[cur][0].v, false, bfr[kb].v, (short)0, acc[0], false, false);
            acc[1] = __builtin_amdgcn_wmma_f32_16x16x32_bf16(
                         false, af[cur][1].v, false, bfr[kb].v, (short)0, acc[1], false, false);
        }

        // Scatter gate tiles to LDS for the elementwise phase.
#pragma unroll
        for (int mm = 0; mm < 2; ++mm) {
            int m = mh * 2 + mm;
#pragma unroll
            for (int r = 0; r < 8; ++r) {
                int row = m * 16 + lhalf * 8 + r;
                gates[q * 1024 + row * 16 + lrow] = acc[mm][r];
            }
        }
        __syncthreads();

        // Elementwise: LSTM cell + zoneout + duration reset masks.
        unsigned char rflag = resetflag[t * BDIM + eb];
        float vmask = (t < mel) ? 1.0f : 0.0f;
        float m     = (!rflag) ? vmask : 0.0f;
        v4f hz4;
        union { __bf16 h[4]; uint2 u; } pk;
#pragma unroll
        for (int jj = 0; jj < 4; ++jj) {
            int jl = j0 + jj;
            float iv = gates[0 * 1024 + eb * 16 + jl];
            float fv = gates[1 * 1024 + eb * 16 + jl];
            float gv = gates[2 * 1024 + eb * 16 + jl];
            float ov = gates[3 * 1024 + eb * 16 + jl];
            iv = 1.0f / (1.0f + __expf(-iv));
            fv = 1.0f / (1.0f + __expf(-fv));
            ov = 1.0f / (1.0f + __expf(-ov));
            float eg = __expf(2.0f * gv); gv = (eg - 1.0f) / (eg + 1.0f);
            float hp = hstate[eb * 16 + jl];
            float cp = cstate[eb * 16 + jl];
            float cn = fv * cp + iv * gv;
            float ec = __expf(2.0f * cn); float tc = (ec - 1.0f) / (ec + 1.0f);
            float hn = ov * tc;
            float h_z = 0.1f * hp + 0.9f * hn;
            float c_z = 0.1f * cp + 0.9f * cn;
            float hm = h_z * m;
            hstate[eb * 16 + jl] = hm;
            cstate[eb * 16 + jl] = c_z * m;
            hz4[jj] = h_z * vmask;
            pk.h[jj] = (__bf16)hm;
        }
        __builtin_nontemporal_store(hz4,
            (v4f*)(outputs + (size_t)t * (BDIM * HDIM) + eb * HDIM + hc0 + j0));
        *(uint2*)(hnxt + eb * HDIM + hc0 + j0) = pk.u;

        // Grid-wide barrier (release hnxt stores, then arrive + tight spin).
        __threadfence();
        __syncthreads();
        if (tid == 0) {
            __hip_atomic_fetch_add(counter, 1, __ATOMIC_RELEASE, __HIP_MEMORY_SCOPE_AGENT);
            int target = NBLK * (t + 1);
            while (__hip_atomic_load(counter, __ATOMIC_ACQUIRE, __HIP_MEMORY_SCOPE_AGENT) < target) {}
        }
        __syncthreads();
    }

    // Final carry (masked) h, c.
    v4f hfin, cfin;
#pragma unroll
    for (int jj = 0; jj < 4; ++jj) {
        hfin[jj] = hstate[eb * 16 + j0 + jj];
        cfin[jj] = cstate[eb * 16 + j0 + jj];
    }
    *(v4f*)(h_out + eb * HDIM + hc0 + j0) = hfin;
    *(v4f*)(c_out + eb * HDIM + hc0 + j0) = cfin;
}

// ---------------------------------------------------------------------------
// Phase-3: segout[b,s,:] = outputs[reset_idx[b,s], b, :]
__global__ void segout_k(const float* __restrict__ outputs,
                         const int* __restrict__ reset_idx,
                         float* __restrict__ seg, int total)
{
    int idx = blockIdx.x * 256 + threadIdx.x;
    if (idx >= total) return;
    int jj = idx & (HDIM - 1);
    int s  = (idx >> 9) & (NSEG - 1);
    int b  = idx >> 16;                    // 512*128 = 65536 per batch
    int t  = reset_idx[b * NSEG + s];
    seg[idx] = outputs[((size_t)t * BDIM + b) * HDIM + jj];
}

// ---------------------------------------------------------------------------
extern "C" void kernel_launch(void* const* d_in, const int* in_sizes, int n_in,
                              void* d_out, int out_size, void* d_ws, size_t ws_size,
                              hipStream_t stream)
{
    const float* x    = (const float*)d_in[0];
    const int*   durs = (const int*)d_in[1];
    const float* Wih  = (const float*)d_in[2];
    const float* Whh  = (const float*)d_in[3];
    const float* bih  = (const float*)d_in[4];
    const float* bhh  = (const float*)d_in[5];
    int T = in_sizes[0] / (BDIM * HDIM);

    char* ws = (char*)d_ws;
    size_t off = 0;
    auto alloc = [&](size_t bytes) {
        size_t p = off; off += (bytes + 255) & ~(size_t)255; return p;
    };
    float*          xg        = (float*)(ws + alloc((size_t)T * BDIM * GDIM * 4));
    unsigned short* WihB      = (unsigned short*)(ws + alloc((size_t)GDIM * HDIM * 2));
    unsigned short* WhhB      = (unsigned short*)(ws + alloc((size_t)GDIM * HDIM * 2));
    float*          biassum   = (float*)(ws + alloc(GDIM * 4));
    unsigned short* hbuf      = (unsigned short*)(ws + alloc(2 * BDIM * HDIM * 2));
    int*            reset_idx = (int*)(ws + alloc(BDIM * NSEG * 4));
    int*            mel_len   = (int*)(ws + alloc(BDIM * 4));
    unsigned char*  resetflag = (unsigned char*)(ws + alloc((size_t)T * BDIM));
    int*            counter   = (int*)(ws + alloc(4));

    float* outputs = (float*)d_out;
    float* h_out   = outputs + (size_t)T * BDIM * HDIM;
    float* c_out   = h_out + BDIM * HDIM;
    float* seg     = c_out + BDIM * HDIM;

    (void)hipMemsetAsync(hbuf, 0, 2 * BDIM * HDIM * 2, stream);
    (void)hipMemsetAsync(resetflag, 0, (size_t)T * BDIM, stream);
    (void)hipMemsetAsync(counter, 0, 4, stream);

    prep_k<<<1, 64, 0, stream>>>(durs, reset_idx, mel_len, resetflag);
    cvt_k<<<(GDIM * HDIM) / 256, 256, 0, stream>>>(Wih, Whh, bih, bhh, WihB, WhhB, biassum);
    gemm_xg_k<<<T * 16, 256, 0, stream>>>(x, WihB, biassum, xg);
    seg_lstm_k<<<NBLK, 256, 0, stream>>>(xg, WhhB, hbuf, resetflag, mel_len,
                                         outputs, h_out, c_out, counter, T);
    int total = BDIM * NSEG * HDIM;
    segout_k<<<total / 256, 256, 0, stream>>>(outputs, reset_idx, seg, total);
}
